// TriAttention_64072322121890
// MI455X (gfx1250) — matrix operand order repaired and checked by reading, not compiled, not run on hardware
//
#include <hip/hip_runtime.h>
#include <hip/hip_bf16.h>
#include <math.h>

typedef __attribute__((ext_vector_type(16))) _Float16 v16h;
typedef __attribute__((ext_vector_type(8)))  _Float16 v8h;
typedef __attribute__((ext_vector_type(8)))  float    v8f;

#define B_   16
#define N_   96
#define D_   256
#define H_   128
#define G_   2
#define VQ_  (N_*N_)          // 9216
#define LROW_ (N_*G_)         // 192 : (a*2+g)
#define LBATCH_ (VQ_*LROW_)   // 1769472 logits per batch

// ---------------------------------------------------------------- helpers
__device__ inline void atomicMaxF(float* addr, float val) {
    if (val >= 0.f) atomicMax((int*)addr, __float_as_int(val));
    else            atomicMin((unsigned int*)addr, (unsigned int)__float_as_int(val));
}

// ---------------------------------------------------------------- small prep kernels
__global__ void k_init_stats(float* gmax, float* gsum) {
    int t = threadIdx.x;
    if (t < B_*G_) { gmax[t] = -INFINITY; gsum[t] = 0.f; }
}

__global__ void k_mask(const float* __restrict__ v, float* __restrict__ maskb) {
    int t = blockIdx.x*blockDim.x + threadIdx.x;
    if (t >= B_*N_) return;
    const float* row = v + (size_t)t * D_;
    float s = 0.f;
    for (int d = 0; d < D_; ++d) s += fabsf(row[d]);
    maskb[t] = (s == 0.f) ? 1.f : 0.f;
}

// f32 -> f16 copy
__global__ void k_cvt16(const float* __restrict__ X, _Float16* __restrict__ Xh, int n) {
    int t = blockIdx.x*256 + threadIdx.x;
    if (t < n) Xh[t] = (_Float16)X[t];
}

// W[K,N] f32 row-major  ->  Wt[N,K] f16 (i.e. column of W contiguous)
__global__ void k_wt(const float* __restrict__ W, _Float16* __restrict__ Wt, int K, int N) {
    int t = blockIdx.x*256 + threadIdx.x;
    if (t >= K*N) return;
    int k = t % K, n = t / K;
    Wt[t] = (_Float16)W[(size_t)k*N + n];
}

// ---------------------------------------------------------------- unified WMMA GEMM
// C[M,N] (f32, ldc=N) = act(A[M,K]f16 @ B[K,N] + bias) (+resid[M,N]f32)
// B supplied transposed: Bt[N,K] f16.  One wave per 16x16*NT tiles, 4 waves/block.
// K templated & fully unrolled; RELU/RESID/BIAS templated for branch-free epilogue.
template<int K, int NT, bool RELU, bool RESID, bool BIAS>
__global__ void __launch_bounds__(128)
k_gemm(const _Float16* __restrict__ A, const _Float16* __restrict__ Bt,
       const float* __restrict__ bias, const float* __restrict__ resid,
       float* __restrict__ C, int M, int N,
       size_t sA, size_t sB, size_t sC)
{
    int z = blockIdx.z;
    A  += (size_t)z * sA;
    Bt += (size_t)z * sB;
    C  += (size_t)z * sC;
    int lane = threadIdx.x;                       // wave32 lane
    int r = lane & 15, hs = lane >> 4;
    int tm = blockIdx.y;
    int tnBase = (blockIdx.x * 4 + threadIdx.y) * NT;   // tile units
    if (tnBase * 16 >= N) return;                 // uniform per wave
    int row = tm*16 + r;
    const _Float16* Arow = A + (size_t)row * K;

    v8f acc[NT];
#pragma unroll
    for (int j = 0; j < NT; ++j) acc[j] = (v8f){};

#pragma unroll
    for (int k0 = 0; k0 < K; k0 += 32) {
        // A 16x32 frag: lanes<16 hold K{0..15}, lanes>=16 K{16..31}
        v8h a0 = *(const v8h*)(Arow + k0 + 16*hs);
        v8h a1 = *(const v8h*)(Arow + k0 + 16*hs + 8);
        v16h af;
#pragma unroll
        for (int e = 0; e < 8; ++e) { af[e] = a0[e]; af[e+8] = a1[e]; }
#pragma unroll
        for (int j = 0; j < NT; ++j) {
            int col = (tnBase + j)*16 + r;
            // B 32x16 frag: contiguous 16 halves from transposed storage
            v16h bf = *(const v16h*)(Bt + (size_t)col*K + k0 + 16*hs);
            acc[j] = __builtin_amdgcn_wmma_f32_16x16x32_f16(false, af, false, bf,
                                                            (short)0, acc[j], false, false);
        }
    }
#pragma unroll
    for (int j = 0; j < NT; ++j) {
        int n = (tnBase + j)*16 + r;
#pragma unroll
        for (int i = 0; i < 8; ++i) {             // C/D: VGPR i -> M = i + 8*hs, N = lane&15
            int m = tm*16 + i + 8*hs;
            float x = acc[j][i];
            if (BIAS)  x += bias[n];
            if (RELU)  x = fmaxf(x, 0.f);
            if (RESID) x += resid[(size_t)m*N + n];
            C[(size_t)m*N + n] = x;
        }
    }
}

// ---------------------------------------------------------------- logits operand prep
__global__ void k_prep_x(const float* __restrict__ vt, const float* __restrict__ qt,
                         _Float16* __restrict__ X) {
    int t = blockIdx.x*256 + threadIdx.x;
    if (t >= B_*VQ_*H_) return;
    int h  = t % H_;
    int vq = (t / H_) % VQ_;
    int b  = t / (H_*VQ_);
    int v = vq / N_, q = vq % N_;
    X[t] = (_Float16)(vt[((size_t)(b*N_+v))*H_ + h] * qt[((size_t)(b*N_+q))*H_ + h]);
}

// ATGt[b][n][h] = at[b,a,h] * Wg[h,g], n = a*2+g  (transposed: h contiguous)
__global__ void k_prep_atg(const float* __restrict__ at, const float* __restrict__ Wg,
                           _Float16* __restrict__ ATGt) {
    int t = blockIdx.x*256 + threadIdx.x;
    if (t >= B_*LROW_*H_) return;
    int h = t % H_;
    int n = (t / H_) % LROW_;
    int b = t / (H_*LROW_);
    int a = n >> 1, g = n & 1;
    ATGt[t] = (_Float16)(at[((size_t)(b*N_+a))*H_ + h] * Wg[h*G_ + g]);
}

// ---------------------------------------------------------------- softmax pass 1: per-(b,g) max
__global__ void k_logit_max(const float* __restrict__ Lg, const float* __restrict__ maskb,
                            float* __restrict__ gmax) {
    int b = blockIdx.y;
    int i = blockIdx.x*256 + threadIdx.x;   // vqa index (grid sized exactly)
    int v = i / (N_*N_);
    float m0 = -INFINITY, m1 = -INFINITY;
    if (maskb[b*N_ + v] == 0.f) {
        const float* p = Lg + (size_t)b*LBATCH_ + (size_t)i*G_;
        m0 = p[0]; m1 = p[1];
    }
    __shared__ float s0[256], s1[256];
    s0[threadIdx.x] = m0; s1[threadIdx.x] = m1;
    __syncthreads();
    for (int s = 128; s > 0; s >>= 1) {
        if (threadIdx.x < s) {
            s0[threadIdx.x] = fmaxf(s0[threadIdx.x], s0[threadIdx.x + s]);
            s1[threadIdx.x] = fmaxf(s1[threadIdx.x], s1[threadIdx.x + s]);
        }
        __syncthreads();
    }
    if (threadIdx.x == 0) {
        atomicMaxF(&gmax[b*G_ + 0], s0[0]);
        atomicMaxF(&gmax[b*G_ + 1], s1[0]);
    }
}

// ---------------------------------------------------------------- softmax pass 2: exp in place + sums
__global__ void k_expsum(float* __restrict__ Lg, const float* __restrict__ maskb,
                         const float* __restrict__ gmax, float* __restrict__ gsum) {
    int b = blockIdx.y;
    int i = blockIdx.x*256 + threadIdx.x;
    int v = i / (N_*N_);
    float* p = Lg + (size_t)b*LBATCH_ + (size_t)i*G_;
    float e0 = 0.f, e1 = 0.f;
    if (maskb[b*N_ + v] == 0.f) {
        e0 = expf(p[0] - gmax[b*G_ + 0]);
        e1 = expf(p[1] - gmax[b*G_ + 1]);
    }
    p[0] = e0; p[1] = e1;                   // Lg now holds unnormalized p
    __shared__ float s0[256], s1[256];
    s0[threadIdx.x] = e0; s1[threadIdx.x] = e1;
    __syncthreads();
    for (int s = 128; s > 0; s >>= 1) {
        if (threadIdx.x < s) {
            s0[threadIdx.x] += s0[threadIdx.x + s];
            s1[threadIdx.x] += s1[threadIdx.x + s];
        }
        __syncthreads();
    }
    if (threadIdx.x == 0) {
        atomicAdd(&gsum[b*G_ + 0], s0[0]);
        atomicAdd(&gsum[b*G_ + 1], s1[0]);
    }
}

// ---------------------------------------------------------------- marginals of p (normalized)
__global__ void k_pvq(const float* __restrict__ Pu, const float* __restrict__ gsum,
                      float* __restrict__ Pvq) {
    int t = blockIdx.x*256 + threadIdx.x;
    if (t >= B_*N_*N_*G_) return;
    int g = t & 1;
    int q = (t >> 1) % N_;
    int v = (t / (G_*N_)) % N_;
    int b = t / (G_*N_*N_);
    const float* p = Pu + (size_t)b*LBATCH_ + (size_t)(v*N_ + q)*LROW_ + g;
    float s = 0.f;
    for (int a = 0; a < N_; ++a) s += p[a*G_];
    Pvq[t] = s / gsum[b*G_ + g];
}

__global__ void k_pva(const float* __restrict__ Pu, const float* __restrict__ gsum,
                      float* __restrict__ Pva) {
    int t = blockIdx.x*256 + threadIdx.x;
    if (t >= B_*N_*N_*G_) return;
    int g = t & 1;
    int a = (t >> 1) % N_;
    int v = (t / (G_*N_)) % N_;
    int b = t / (G_*N_*N_);
    const float* p = Pu + (size_t)b*LBATCH_ + (size_t)v*N_*LROW_ + a*G_ + g;
    float s = 0.f;
    for (int q = 0; q < N_; ++q) s += p[(size_t)q*LROW_];
    Pva[t] = s / gsum[b*G_ + g];
}

__global__ void k_pqa(const float* __restrict__ Pu, const float* __restrict__ gsum,
                      float* __restrict__ Pqa) {
    int t = blockIdx.x*256 + threadIdx.x;
    if (t >= B_*N_*N_*G_) return;
    int g = t & 1;
    int a = (t >> 1) % N_;
    int q = (t / (G_*N_)) % N_;
    int b = t / (G_*N_*N_);
    const float* p = Pu + (size_t)b*LBATCH_ + (size_t)q*LROW_ + a*G_ + g;
    float s = 0.f;
    for (int v = 0; v < N_; ++v) s += p[(size_t)v*N_*LROW_];
    Pqa[t] = s / gsum[b*G_ + g];
}

// ---------------------------------------------------------------- attended contractions -> f16 operands
__global__ void k_cv(const float* __restrict__ Pvq, const float* __restrict__ Pva,
                     const float* __restrict__ qp, const float* __restrict__ ap,
                     _Float16* __restrict__ Cv) {
    int t = blockIdx.x*256 + threadIdx.x;
    if (t >= B_*N_*G_*H_) return;
    int d = t % H_;
    int g = (t / H_) & 1;
    int v = (t / (H_*G_)) % N_;
    int b = t / (H_*G_*N_);
    const float* pq = Pvq + (size_t)(b*N_ + v)*N_*G_ + g;
    const float* pa = Pva + (size_t)(b*N_ + v)*N_*G_ + g;
    float s = 0.f;
    for (int q = 0; q < N_; ++q) s += pq[q*G_] * qp[((size_t)(b*N_+q))*H_ + d];
    for (int a = 0; a < N_; ++a) s += pa[a*G_] * ap[((size_t)(b*N_+a))*H_ + d];
    Cv[((size_t)(b*N_+v))*(G_*H_) + g*H_ + d] = (_Float16)s;   // [b,v, g*128+d]
}

__global__ void k_cq(const float* __restrict__ Pvq, const float* __restrict__ Pqa,
                     const float* __restrict__ vp, const float* __restrict__ ap,
                     _Float16* __restrict__ Cq) {
    int t = blockIdx.x*256 + threadIdx.x;
    if (t >= B_*N_*G_*H_) return;
    int d = t % H_;
    int g = (t / H_) & 1;
    int q = (t / (H_*G_)) % N_;
    int b = t / (H_*G_*N_);
    const float* pv = Pvq + ((size_t)b*N_*N_ + q)*G_ + g;    // stride over v = N*G
    const float* pa = Pqa + (size_t)(b*N_ + q)*N_*G_ + g;    // stride over a = G
    float s = 0.f;
    for (int v = 0; v < N_; ++v) s += pv[(size_t)v*N_*G_] * vp[((size_t)(b*N_+v))*H_ + d];
    for (int a = 0; a < N_; ++a) s += pa[a*G_] * ap[((size_t)(b*N_+a))*H_ + d];
    Cq[((size_t)(b*N_+q))*(G_*H_) + g*H_ + d] = (_Float16)s; // [b,q, g*128+d]
}

__global__ void k_ca(const float* __restrict__ Pva, const float* __restrict__ Pqa,
                     const float* __restrict__ vp, const float* __restrict__ qp,
                     _Float16* __restrict__ Ca) {
    int t = blockIdx.x*256 + threadIdx.x;
    if (t >= B_*N_*G_*H_) return;
    int d = t % H_;
    int g = (t / H_) & 1;
    int a = (t / (H_*G_)) % N_;
    int b = t / (H_*G_*N_);
    const float* pv = Pva + ((size_t)b*N_*N_ + a)*G_ + g;    // stride over v = N*G
    const float* pq = Pqa + ((size_t)b*N_*N_ + a)*G_ + g;    // stride over q = N*G
    float s = 0.f;
    for (int v = 0; v < N_; ++v) s += pv[(size_t)v*N_*G_] * vp[((size_t)(b*N_+v))*H_ + d];
    for (int q = 0; q < N_; ++q) s += pq[(size_t)q*N_*G_] * qp[((size_t)(b*N_+q))*H_ + d];
    Ca[((size_t)(b*N_+a))*(G_*H_) + d*G_ + g] = (_Float16)s; // [b,a, d*2+g] per reference
}

// ---------------------------------------------------------------- launcher
extern "C" void kernel_launch(void* const* d_in, const int* in_sizes, int n_in,
                              void* d_out, int out_size, void* d_ws, size_t ws_size,
                              hipStream_t stream) {
    const float* v   = (const float*)d_in[0];
    const float* q   = (const float*)d_in[1];
    const float* a   = (const float*)d_in[2];
    const float* Wvt = (const float*)d_in[3];  const float* bvt = (const float*)d_in[4];
    const float* Wqt = (const float*)d_in[5];  const float* bqt = (const float*)d_in[6];
    const float* Wat = (const float*)d_in[7];  const float* bat = (const float*)d_in[8];
    const float* Wg  = (const float*)d_in[9];
    const float* Wvp = (const float*)d_in[10]; const float* bvp = (const float*)d_in[11];
    const float* Wqp = (const float*)d_in[12]; const float* bqp = (const float*)d_in[13];
    const float* Wap = (const float*)d_in[14]; const float* bap = (const float*)d_in[15];
    const float* Wvu = (const float*)d_in[16]; const float* bvu = (const float*)d_in[17];
    const float* Wqu = (const float*)d_in[18]; const float* bqu = (const float*)d_in[19];
    const float* Wau = (const float*)d_in[20]; const float* bau = (const float*)d_in[21];
    float* out = (float*)d_out;
    (void)ws_size; (void)in_sizes; (void)n_in; (void)out_size;

    // workspace carve-up (float units; every region a multiple of 8 floats -> 32B aligned)
    float* w = (float*)d_ws;
    size_t o = 0;
    float* vt = w + o; o += (size_t)B_*N_*H_;         // f32 activations
    float* qt = w + o; o += (size_t)B_*N_*H_;
    float* at = w + o; o += (size_t)B_*N_*H_;
    float* vp = w + o; o += (size_t)B_*N_*H_;
    float* qp = w + o; o += (size_t)B_*N_*H_;
    float* ap = w + o; o += (size_t)B_*N_*H_;
    float* maskb = w + o; o += (size_t)B_*N_;
    float* gmax  = w + o; o += 32;
    float* gsum  = w + o; o += 32;
    float* Pvq = w + o; o += (size_t)B_*N_*N_*G_;
    float* Pva = w + o; o += (size_t)B_*N_*N_*G_;
    float* Pqa = w + o; o += (size_t)B_*N_*N_*G_;
    _Float16* vh = (_Float16*)(w + o); o += (size_t)B_*N_*D_/2;    // f16 inputs
    _Float16* qh = (_Float16*)(w + o); o += (size_t)B_*N_*D_/2;
    _Float16* ah = (_Float16*)(w + o); o += (size_t)B_*N_*D_/2;
    _Float16* WvtT = (_Float16*)(w + o); o += (size_t)D_*H_/2;     // f16 transposed weights [N,K]
    _Float16* WqtT = (_Float16*)(w + o); o += (size_t)D_*H_/2;
    _Float16* WatT = (_Float16*)(w + o); o += (size_t)D_*H_/2;
    _Float16* WvpT = (_Float16*)(w + o); o += (size_t)D_*H_/2;
    _Float16* WqpT = (_Float16*)(w + o); o += (size_t)D_*H_/2;
    _Float16* WapT = (_Float16*)(w + o); o += (size_t)D_*H_/2;
    _Float16* WvuT = (_Float16*)(w + o); o += (size_t)D_*D_/2;
    _Float16* WquT = (_Float16*)(w + o); o += (size_t)D_*D_/2;
    _Float16* WauT = (_Float16*)(w + o); o += (size_t)D_*D_/2;
    _Float16* Chv = (_Float16*)(w + o); o += (size_t)B_*N_*D_/2;   // f16 glimpse features
    _Float16* Chq = (_Float16*)(w + o); o += (size_t)B_*N_*D_/2;
    _Float16* Cha = (_Float16*)(w + o); o += (size_t)B_*N_*D_/2;
    _Float16* Xh   = (_Float16*)(w + o); o += (size_t)B_*VQ_*H_/2; // logits A operand
    _Float16* ATGt = (_Float16*)(w + o); o += (size_t)B_*LROW_*H_/2;
    float* Lg = w + o; o += (size_t)B_*LBATCH_;                    // logits -> unnormalized p

    const int M = B_*N_;              // 1536
    dim3 blk(32, 4);

    // 0) f16 conversions: inputs + 9 transposed weights
    int nin = M*D_;
    k_cvt16<<<(nin+255)/256, 256, 0, stream>>>(v, vh, nin);
    k_cvt16<<<(nin+255)/256, 256, 0, stream>>>(q, qh, nin);
    k_cvt16<<<(nin+255)/256, 256, 0, stream>>>(a, ah, nin);
    k_wt<<<(D_*H_+255)/256, 256, 0, stream>>>(Wvt, WvtT, D_, H_);
    k_wt<<<(D_*H_+255)/256, 256, 0, stream>>>(Wqt, WqtT, D_, H_);
    k_wt<<<(D_*H_+255)/256, 256, 0, stream>>>(Wat, WatT, D_, H_);
    k_wt<<<(D_*H_+255)/256, 256, 0, stream>>>(Wvp, WvpT, D_, H_);
    k_wt<<<(D_*H_+255)/256, 256, 0, stream>>>(Wqp, WqpT, D_, H_);
    k_wt<<<(D_*H_+255)/256, 256, 0, stream>>>(Wap, WapT, D_, H_);
    k_wt<<<(D_*D_+255)/256, 256, 0, stream>>>(Wvu, WvuT, D_, D_);
    k_wt<<<(D_*D_+255)/256, 256, 0, stream>>>(Wqu, WquT, D_, D_);
    k_wt<<<(D_*D_+255)/256, 256, 0, stream>>>(Wau, WauT, D_, D_);

    // 1) six projection GEMMs (WMMA, K=256, NT=2): N=128 -> 8 tiles -> grid.x=1
    dim3 gProj(1, M/16, 1);
    k_gemm<D_,2,true ,false,true><<<gProj, blk, 0, stream>>>(vh, WvtT, bvt, nullptr, vt, M, H_, 0,0,0);
    k_gemm<D_,2,true ,false,true><<<gProj, blk, 0, stream>>>(qh, WqtT, bqt, nullptr, qt, M, H_, 0,0,0);
    k_gemm<D_,2,true ,false,true><<<gProj, blk, 0, stream>>>(ah, WatT, bat, nullptr, at, M, H_, 0,0,0);
    k_gemm<D_,2,false,false,true><<<gProj, blk, 0, stream>>>(vh, WvpT, bvp, nullptr, vp, M, H_, 0,0,0);
    k_gemm<D_,2,false,false,true><<<gProj, blk, 0, stream>>>(qh, WqpT, bqp, nullptr, qp, M, H_, 0,0,0);
    k_gemm<D_,2,false,false,true><<<gProj, blk, 0, stream>>>(ah, WapT, bap, nullptr, ap, M, H_, 0,0,0);

    // 2) logits operands, mask, stats init
    k_prep_x  <<<(B_*VQ_*H_ + 255)/256, 256, 0, stream>>>(vt, qt, Xh);
    k_prep_atg<<<(B_*LROW_*H_ + 255)/256, 256, 0, stream>>>(at, Wg, ATGt);
    k_mask    <<<(B_*N_ + 255)/256, 256, 0, stream>>>(v, maskb);
    k_init_stats<<<1, 32, 0, stream>>>(gmax, gsum);

    // 3) batched logits GEMM (WMMA, K=128, NT=3): 12 N-tiles = 4 waves * 3
    dim3 gL(1, VQ_/16, B_);
    k_gemm<H_,3,false,false,false><<<gL, blk, 0, stream>>>(
        Xh, ATGt, nullptr, nullptr, Lg, VQ_, LROW_,
        (size_t)VQ_*H_, (size_t)LROW_*H_, (size_t)LBATCH_);

    // 4) softmax over (v,q,a) per (b,g): max, then exp-in-place + sum
    dim3 gS((N_*N_*N_)/256, B_);
    k_logit_max<<<gS, 256, 0, stream>>>(Lg, maskb, gmax);
    k_expsum  <<<gS, 256, 0, stream>>>(Lg, maskb, gmax, gsum);

    // 5) pairwise marginals of p (downstream math only needs these)
    int nM = B_*N_*N_*G_;
    k_pvq<<<(nM + 255)/256, 256, 0, stream>>>(Lg, gsum, Pvq);
    k_pva<<<(nM + 255)/256, 256, 0, stream>>>(Lg, gsum, Pva);
    k_pqa<<<(nM + 255)/256, 256, 0, stream>>>(Lg, gsum, Pqa);

    // 6) attended contractions -> flattened f16 glimpse features
    int nC = B_*N_*G_*H_;
    k_cv<<<(nC + 255)/256, 256, 0, stream>>>(Pvq, Pva, qp, ap, Chv);
    k_cq<<<(nC + 255)/256, 256, 0, stream>>>(Pvq, Pqa, vp, ap, Chq);
    k_ca<<<(nC + 255)/256, 256, 0, stream>>>(Pva, Pqa, vp, qp, Cha);

    // 7) residual update GEMMs (WMMA, K=256, NT=2): N=256 -> 16 tiles -> grid.x=2
    dim3 gU(2, M/16, 1);
    k_gemm<D_,2,false,true,true><<<gU, blk, 0, stream>>>(Chv, WvuT, bvu, v, out,                 M, D_, 0,0,0);
    k_gemm<D_,2,false,true,true><<<gU, blk, 0, stream>>>(Chq, WquT, bqu, q, out + (size_t)M*D_,  M, D_, 0,0,0);
    k_gemm<D_,2,false,true,true><<<gU, blk, 0, stream>>>(Cha, WauT, bau, a, out + (size_t)2*M*D_, M, D_, 0,0,0);
}
